// PerResidueLDDTCaPredictor_11708080849376
// MI455X (gfx1250) — compile-verified, matrix operand order checked by reading, not run on hardware
//
#include <hip/hip_runtime.h>
#include <hip/hip_bf16.h>
#include <math.h>

#define L_SEQ 768
#define C_IN  1024
#define CH    128
#define NHEAD 8
#define HDIM  16
#define NB    50
#define NBPAD 64
#define PDE_O 64
#define ATT_SCALE 0.25f   /* 1/sqrt(16) */

typedef __attribute__((ext_vector_type(16))) __bf16        v16bf;
typedef __attribute__((ext_vector_type(8)))  float         v8f;
typedef __attribute__((ext_vector_type(4)))  unsigned int  v4u;

// ---------------------------------------------------------------- helpers
__device__ __forceinline__ unsigned short f2bf(float x) {
  union { float f; unsigned int u; } c; c.f = x;
  unsigned int u = c.u;
  return (unsigned short)((u + 0x7FFFu + ((u >> 16) & 1u)) >> 16);
}
__device__ __forceinline__ float silu_f(float x) { return x / (1.f + __expf(-x)); }
__device__ __forceinline__ float gelu_f(float x) {
  float x3 = x * x * x;
  return 0.5f * x * (1.f + tanhf(0.7978845608028654f * (x + 0.044715f * x3)));
}

// Fragment load, row-major source [R x K], stride ld (elements, mult of 8).
// Lane holds row r0+lane%16; K-chunks {half*8..+7} and {16+half*8..+7}:
// exactly two contiguous 16-byte loads (b128). Used for A ([M x K]) and
// for B^T ([N x K]); works for global or LDS pointers.
__device__ __forceinline__ v16bf load_frag(const unsigned short* __restrict__ p,
                                           int ld, int r0, int k0) {
  const int lane = threadIdx.x & 31;
  const int half = lane >> 4;
  const unsigned short* base =
      p + (size_t)(r0 + (lane & 15)) * ld + k0 + half * 8;
  union { v4u u[2]; v16bf f; } cv;
  cv.u[0] = *(const v4u*)(base);
  cv.u[1] = *(const v4u*)(base + 16);
  return cv.f;
}

// ---------------------------------------------------------------- GEMM
// C[M,N] = epi(A[M,K] @ Bt[N,K]^T).  M multiple of 128; K multiple of 32.
// Epilogues: 0 plain(+bias), 1 SiLU, 2 GELU(tanh),
//            3 gated residual: C = aux + gate*(acc+bias),
//            4 scores: C = acc*alpha + aux2d
template <int EPI, int NT>
__global__ __launch_bounds__(256) void gemm_wmma(
    const unsigned short* __restrict__ A, int lda,
    const unsigned short* __restrict__ Bt, int ldb,
    float* __restrict__ C, int ldc, unsigned short* __restrict__ Cbf,
    const float* __restrict__ bias,
    const float* __restrict__ aux, int ldaux,
    const float* __restrict__ gate, int ldgate,
    int N, int K, float alpha) {
  const int lane = threadIdx.x & 31;
  const int wid  = threadIdx.x >> 5;
  const int row0 = blockIdx.y * 128 + wid * 16;
  const int col0 = blockIdx.x * (NT * 16);

  v8f acc[NT] = {};
  for (int k0 = 0; k0 < K; k0 += 32) {
    v16bf a = load_frag(A, lda, row0, k0);
#pragma unroll
    for (int j = 0; j < NT; ++j) {
      v16bf b = load_frag(Bt, ldb, col0 + j * 16, k0);
      acc[j] = __builtin_amdgcn_wmma_f32_16x16x32_bf16(
          false, a, false, b, (short)0, acc[j], false, false);
    }
  }
#pragma unroll
  for (int j = 0; j < NT; ++j) {
    const int col = col0 + j * 16 + (lane & 15);
    if (col >= N) continue;
    const int rbase = row0 + 8 * (lane >> 4);
#pragma unroll
    for (int r = 0; r < 8; ++r) {
      const int row = rbase + r;
      float v = acc[j][r] * alpha;
      if (bias) v += bias[col];
      if (EPI == 1) v = silu_f(v);
      else if (EPI == 2) v = gelu_f(v);
      else if (EPI == 3) v = aux[(size_t)row * ldaux + col] +
                             gate[(size_t)row * ldgate + col] * v;
      else if (EPI == 4) v += aux[(size_t)row * ldaux + col];
      if (C)   C[(size_t)row * ldc + col] = v;
      if (Cbf) Cbf[(size_t)row * ldc + col] = f2bf(v);
    }
  }
}

// ---------------------------------------------------------------- LN + adaLN modulation -> bf16
__global__ __launch_bounds__(256) void ln_mod_kernel(
    const float* __restrict__ s, const float* __restrict__ mods,
    int sh_off, int sc_off, unsigned short* __restrict__ out) {
  const int row  = blockIdx.x * 8 + (threadIdx.x >> 5);
  const int lane = threadIdx.x & 31;
  const float4 x4 = ((const float4*)(s + (size_t)row * CH))[lane];
  float x[4] = {x4.x, x4.y, x4.z, x4.w};
  float sum = x[0] + x[1] + x[2] + x[3];
#pragma unroll
  for (int o = 16; o > 0; o >>= 1) sum += __shfl_xor(sum, o, 32);
  const float mean = sum * (1.f / CH);
  float var = 0.f;
#pragma unroll
  for (int j = 0; j < 4; ++j) { float d = x[j] - mean; var += d * d; }
#pragma unroll
  for (int o = 16; o > 0; o >>= 1) var += __shfl_xor(var, o, 32);
  const float rs = rsqrtf(var * (1.f / CH) + 1e-5f);
  const float* mrow = mods + (size_t)row * (6 * CH);
  unsigned int pk[2];
#pragma unroll
  for (int j = 0; j < 4; ++j) {
    const int c = lane * 4 + j;
    const float sc = mrow[sc_off + c];
    const float sh = mrow[sh_off + c];
    const unsigned short b = f2bf((x[j] - mean) * rs * (1.f + sc) + sh);
    if (j & 1) pk[j >> 1] |= (unsigned int)b << 16;
    else       pk[j >> 1] = b;
  }
  ((unsigned int*)(out + (size_t)row * CH))[lane * 2 + 0] = pk[0];
  ((unsigned int*)(out + (size_t)row * CH))[lane * 2 + 1] = pk[1];
}

// ---------------------------------------------------------------- pairwise distance bias (+mask folded in)
__global__ __launch_bounds__(256) void bias2d_kernel(
    const float* __restrict__ pos, const unsigned char* __restrict__ pmask,
    const float* __restrict__ w1, const float* __restrict__ b1,
    const float* __restrict__ w2, const float* __restrict__ b2,
    float* __restrict__ out) {
  __shared__ float sw1[CH], sb1[CH], sw2[CH];
  if (threadIdx.x < CH) {
    sw1[threadIdx.x] = w1[threadIdx.x];
    sb1[threadIdx.x] = b1[threadIdx.x];
    sw2[threadIdx.x] = w2[threadIdx.x];
  }
  __syncthreads();
  const int idx = blockIdx.x * 256 + threadIdx.x;
  const int l = idx / L_SEQ, m = idx % L_SEQ;
  const float dx = pos[l * 3 + 0] - pos[m * 3 + 0];
  const float dy = pos[l * 3 + 1] - pos[m * 3 + 1];
  const float dz = pos[l * 3 + 2] - pos[m * 3 + 2];
  const float d  = sqrtf(dx * dx + dy * dy + dz * dz + 1e-12f);
  float acc = 0.f;
#pragma unroll 4
  for (int c = 0; c < CH; ++c) acc += silu_f(fmaf(d, sw1[c], sb1[c])) * sw2[c];
  acc += b2[0];
  if (pmask[m]) acc += -1e9f;
  out[idx] = acc;
}

// ---------------------------------------------------------------- row softmax (768 wide) -> bf16 P
__global__ __launch_bounds__(256) void softmax_kernel(
    const float* __restrict__ S, unsigned short* __restrict__ P) {
  __shared__ float redm[8], reds[8];
  const int row  = blockIdx.x;
  const int t    = threadIdx.x;
  const int lane = t & 31, wid = t >> 5;
  const float* sr = S + (size_t)row * L_SEQ;
  float x[3];
#pragma unroll
  for (int j = 0; j < 3; ++j) x[j] = sr[t + j * 256];
  float mx = fmaxf(fmaxf(x[0], x[1]), x[2]);
#pragma unroll
  for (int o = 16; o > 0; o >>= 1) mx = fmaxf(mx, __shfl_xor(mx, o, 32));
  if (lane == 0) redm[wid] = mx;
  __syncthreads();
  float gm = redm[0];
#pragma unroll
  for (int i = 1; i < 8; ++i) gm = fmaxf(gm, redm[i]);
  float ss = 0.f;
#pragma unroll
  for (int j = 0; j < 3; ++j) { x[j] = __expf(x[j] - gm); ss += x[j]; }
#pragma unroll
  for (int o = 16; o > 0; o >>= 1) ss += __shfl_xor(ss, o, 32);
  if (lane == 0) reds[wid] = ss;
  __syncthreads();
  float gs = 0.f;
#pragma unroll
  for (int i = 0; i < 8; ++i) gs += reds[i];
  const float inv = 1.f / gs;
#pragma unroll
  for (int j = 0; j < 3; ++j)
    P[(size_t)row * L_SEQ + t + j * 256] = f2bf(x[j] * inv);
}

// ---------------------------------------------------------------- pde head: per-l GEMM [768x128]@[128x64]
__global__ __launch_bounds__(256) void pde_kernel(
    const unsigned short* __restrict__ s_bf, const float* __restrict__ s_f,
    const float* __restrict__ pzw, const float* __restrict__ pzb,
    float* __restrict__ out) {
  __shared__ unsigned short sBt[PDE_O * CH];  // B^T tile [64 x 128], 16 KB
  const int l = blockIdx.x;
  for (int i = threadIdx.x; i < PDE_O * CH; i += 256) {
    const int o = i >> 7, h = i & 127;
    sBt[i] = f2bf(s_f[(size_t)l * CH + h] * pzw[(size_t)h * PDE_O + o]);
  }
  __syncthreads();
  const int lane = threadIdx.x & 31;
  const int wid  = threadIdx.x >> 5;
  float* orow = out + (size_t)l * (L_SEQ * PDE_O);
  for (int t = 0; t < 6; ++t) {
    const int row0 = (wid * 6 + t) * 16;
    v8f acc[4] = {};
#pragma unroll
    for (int k0 = 0; k0 < CH; k0 += 32) {
      v16bf a = load_frag(s_bf, CH, row0, k0);
#pragma unroll
      for (int j = 0; j < 4; ++j) {
        v16bf b = load_frag(sBt, CH, j * 16, k0);
        acc[j] = __builtin_amdgcn_wmma_f32_16x16x32_bf16(
            false, a, false, b, (short)0, acc[j], false, false);
      }
    }
#pragma unroll
    for (int j = 0; j < 4; ++j) {
      const int col = j * 16 + (lane & 15);
      const int rbase = row0 + 8 * (lane >> 4);
#pragma unroll
      for (int r = 0; r < 8; ++r) {
        __builtin_nontemporal_store(acc[j][r] + pzb[col],
                                    &orow[(size_t)(rbase + r) * PDE_O + col]);
      }
    }
  }
}

// ---------------------------------------------------------------- converts / packs
__global__ __launch_bounds__(256) void cvt_bf16x4_kernel(
    const float* __restrict__ in, unsigned int* __restrict__ out, int n4) {
  const int i = blockIdx.x * 256 + threadIdx.x;
  if (i >= n4) return;
  const float4 v = ((const float4*)in)[i];
  out[2 * i + 0] = (unsigned)f2bf(v.x) | ((unsigned)f2bf(v.y) << 16);
  out[2 * i + 1] = (unsigned)f2bf(v.z) | ((unsigned)f2bf(v.w) << 16);
}

// W [K x N] f32 -> W^T bf16 [Npad x K] (rows >= N zero-filled)
__global__ __launch_bounds__(256) void cvt_transpose_kernel(
    const float* __restrict__ in, unsigned short* __restrict__ out,
    int K, int N, int Npad) {
  const int idx = blockIdx.x * 256 + threadIdx.x;
  if (idx >= Npad * K) return;
  const int n = idx / K, k = idx % K;
  out[idx] = (n < N) ? f2bf(in[(size_t)k * N + n]) : (unsigned short)0;
}

// q/k slices of qkv -> per-head [L x 32] zero-padded (K dim 16 -> 32)
__global__ __launch_bounds__(256) void pack_qk_kernel(
    const unsigned short* __restrict__ qkv_bf,
    unsigned short* __restrict__ qp, unsigned short* __restrict__ kp) {
  const int idx = blockIdx.x * 256 + threadIdx.x;  // NHEAD*L*32
  const int d = idx & 31;
  const int lm = idx >> 5;
  const int l = lm % L_SEQ, h = lm / L_SEQ;
  unsigned short q = 0, k = 0;
  if (d < HDIM) {
    q = qkv_bf[(size_t)l * (3 * CH) + h * HDIM + d];
    k = qkv_bf[(size_t)l * (3 * CH) + CH + h * HDIM + d];
  }
  qp[idx] = q;
  kp[idx] = k;
}

// v slice of qkv [L x 128] -> vT [128 x L]
__global__ __launch_bounds__(256) void transpose_v_kernel(
    const unsigned short* __restrict__ qkv_bf, unsigned short* __restrict__ vT) {
  const int idx = blockIdx.x * 256 + threadIdx.x;  // 128*L
  const int c = idx / L_SEQ, m = idx % L_SEQ;
  vT[idx] = qkv_bf[(size_t)m * (3 * CH) + 2 * CH + c];
}

// ---------------------------------------------------------------- host side
static inline void launch_gemm(hipStream_t st, int epi, int nt,
                               const unsigned short* A, int lda,
                               const unsigned short* Bt, int ldb,
                               float* C, int ldc, unsigned short* Cbf,
                               const float* bias,
                               const float* aux, int ldaux,
                               const float* gate, int ldgate,
                               int M, int N, int K, float alpha) {
  dim3 g((N + nt * 16 - 1) / (nt * 16), M / 128);
#define GC(E, T) gemm_wmma<E, T><<<g, 256, 0, st>>>(A, lda, Bt, ldb, C, ldc, Cbf, bias, aux, ldaux, gate, ldgate, N, K, alpha)
  if (nt == 1) {
    GC(0, 1);
  } else {
    switch (epi) {
      case 0: GC(0, 4); break;
      case 1: GC(1, 4); break;
      case 2: GC(2, 4); break;
      case 3: GC(3, 4); break;
      default: GC(4, 4); break;
    }
  }
#undef GC
}

static inline void cvt4(hipStream_t st, const float* in, unsigned short* out, int n) {
  cvt_bf16x4_kernel<<<(n / 4 + 255) / 256, 256, 0, st>>>(in, (unsigned int*)out, n / 4);
}
static inline void cvtT(hipStream_t st, const float* in, unsigned short* out,
                        int K, int N, int Npad) {
  cvt_transpose_kernel<<<(Npad * K + 255) / 256, 256, 0, st>>>(in, out, K, N, Npad);
}

extern "C" void kernel_launch(void* const* d_in, const int* in_sizes, int n_in,
                              void* d_out, int out_size, void* d_ws, size_t ws_size,
                              hipStream_t stream) {
  (void)in_sizes; (void)n_in; (void)out_size; (void)ws_size;
  const float* x_dec   = (const float*)d_in[0];
  const float* enc_out = (const float*)d_in[1];
  const float* pos     = (const float*)d_in[2];
  const unsigned char* pmask = (const unsigned char*)d_in[3];
  const float* ls_w = (const float*)d_in[4];  const float* ls_b = (const float*)d_in[5];
  const float* lc_w = (const float*)d_in[6];  const float* lc_b = (const float*)d_in[7];
  const float* dp1_w = (const float*)d_in[8]; const float* dp1_b = (const float*)d_in[9];
  const float* dp2_w = (const float*)d_in[10]; const float* dp2_b = (const float*)d_in[11];
  const float* adaln_w = (const float*)d_in[12]; const float* adaln_b = (const float*)d_in[13];
  const float* qkv_w = (const float*)d_in[14]; const float* qkv_b = (const float*)d_in[15];
  const float* out_w = (const float*)d_in[16]; const float* out_b = (const float*)d_in[17];
  const float* fc1_w = (const float*)d_in[18]; const float* fc1_b = (const float*)d_in[19];
  const float* fc2_w = (const float*)d_in[20]; const float* fc2_b = (const float*)d_in[21];
  const float* ps_w = (const float*)d_in[22];  const float* ps_b = (const float*)d_in[23];
  const float* pz_w = (const float*)d_in[24];  const float* pz_b = (const float*)d_in[25];

  char* ws = (char*)d_ws;
  size_t off = 0;
  auto alloc = [&](size_t bytes) -> char* {
    char* p = ws + off;
    off += (bytes + 255) & ~(size_t)255;
    return p;
  };

  // bf16 activations (row-major A matrices)
  unsigned short* x_bf  = (unsigned short*)alloc((size_t)L_SEQ * C_IN * 2);
  unsigned short* e_bf  = (unsigned short*)alloc((size_t)L_SEQ * C_IN * 2);
  // bf16 transposed weights [N x K]
  unsigned short* ls_t  = (unsigned short*)alloc((size_t)CH * C_IN * 2);
  unsigned short* lc_t  = (unsigned short*)alloc((size_t)CH * C_IN * 2);
  unsigned short* ada_t = (unsigned short*)alloc((size_t)4 * 6 * CH * CH * 2);
  unsigned short* qw_t  = (unsigned short*)alloc((size_t)4 * 3 * CH * CH * 2);
  unsigned short* ow_t  = (unsigned short*)alloc((size_t)4 * CH * CH * 2);
  unsigned short* f1_t  = (unsigned short*)alloc((size_t)4 * CH * CH * 2);
  unsigned short* f2_t  = (unsigned short*)alloc((size_t)4 * CH * CH * 2);
  unsigned short* ps_t  = (unsigned short*)alloc((size_t)NBPAD * CH * 2);
  // activations
  float* s_f32   = (float*)alloc((size_t)L_SEQ * CH * 4);
  float* cond_f  = (float*)alloc((size_t)L_SEQ * CH * 4);
  unsigned short* cond_bf = (unsigned short*)alloc((size_t)L_SEQ * CH * 2);
  float* mods    = (float*)alloc((size_t)L_SEQ * 6 * CH * 4);
  unsigned short* xn_bf  = (unsigned short*)alloc((size_t)L_SEQ * CH * 2);
  unsigned short* qkv_bf = (unsigned short*)alloc((size_t)L_SEQ * 3 * CH * 2);
  unsigned short* qp     = (unsigned short*)alloc((size_t)NHEAD * L_SEQ * 32 * 2);
  unsigned short* kp     = (unsigned short*)alloc((size_t)NHEAD * L_SEQ * 32 * 2);
  unsigned short* vT     = (unsigned short*)alloc((size_t)CH * L_SEQ * 2);
  float* bias2d  = (float*)alloc((size_t)L_SEQ * L_SEQ * 4);
  float* scores  = (float*)alloc((size_t)NHEAD * L_SEQ * L_SEQ * 4);
  unsigned short* p_bf = (unsigned short*)alloc((size_t)NHEAD * L_SEQ * L_SEQ * 2);
  float* o_f32   = (float*)alloc((size_t)L_SEQ * CH * 4);
  unsigned short* o_bf = (unsigned short*)alloc((size_t)L_SEQ * CH * 2);
  unsigned short* t_bf = (unsigned short*)alloc((size_t)L_SEQ * CH * 2);
  unsigned short* s_bf = (unsigned short*)alloc((size_t)L_SEQ * CH * 2);

  // ---- conversions
  cvt4(stream, x_dec, x_bf, L_SEQ * C_IN);
  cvt4(stream, enc_out, e_bf, L_SEQ * C_IN);
  cvtT(stream, ls_w, ls_t, C_IN, CH, CH);
  cvtT(stream, lc_w, lc_t, C_IN, CH, CH);
  for (int i = 0; i < 4; ++i) {
    cvtT(stream, adaln_w + (size_t)i * CH * 6 * CH, ada_t + (size_t)i * 6 * CH * CH,
         CH, 6 * CH, 6 * CH);
    cvtT(stream, qkv_w + (size_t)i * CH * 3 * CH, qw_t + (size_t)i * 3 * CH * CH,
         CH, 3 * CH, 3 * CH);
    cvtT(stream, out_w + (size_t)i * CH * CH, ow_t + (size_t)i * CH * CH, CH, CH, CH);
    cvtT(stream, fc1_w + (size_t)i * CH * CH, f1_t + (size_t)i * CH * CH, CH, CH, CH);
    cvtT(stream, fc2_w + (size_t)i * CH * CH, f2_t + (size_t)i * CH * CH, CH, CH, CH);
  }
  cvtT(stream, ps_w, ps_t, CH, NB, NBPAD);

  // ---- trunk
  // s = x_dec @ ls_w + ls_b ; cond = silu(enc_out @ lc_w + lc_b)
  launch_gemm(stream, 0, 4, x_bf, C_IN, ls_t, C_IN, s_f32, CH, nullptr,
              ls_b, nullptr, 0, nullptr, 0, L_SEQ, CH, C_IN, 1.f);
  launch_gemm(stream, 1, 4, e_bf, C_IN, lc_t, C_IN, cond_f, CH, cond_bf,
              lc_b, nullptr, 0, nullptr, 0, L_SEQ, CH, C_IN, 1.f);

  bias2d_kernel<<<(L_SEQ * L_SEQ) / 256, 256, 0, stream>>>(
      pos, pmask, dp1_w, dp1_b, dp2_w, dp2_b, bias2d);

  for (int i = 0; i < 4; ++i) {
    // mods = cond @ adaln_w[i] + adaln_b[i]
    launch_gemm(stream, 0, 4, cond_bf, CH, ada_t + (size_t)i * 6 * CH * CH, CH,
                mods, 6 * CH, nullptr, adaln_b + (size_t)i * 6 * CH,
                nullptr, 0, nullptr, 0, L_SEQ, 6 * CH, CH, 1.f);
    // xn = ln(s)*(1+sc_a)+sh_a
    ln_mod_kernel<<<L_SEQ / 8, 256, 0, stream>>>(s_f32, mods, 0, CH, xn_bf);
    // qkv = xn @ qkv_w[i] + qkv_b[i]  (bf16 only)
    launch_gemm(stream, 0, 4, xn_bf, CH, qw_t + (size_t)i * 3 * CH * CH, CH,
                nullptr, 3 * CH, qkv_bf, qkv_b + (size_t)i * 3 * CH,
                nullptr, 0, nullptr, 0, L_SEQ, 3 * CH, CH, 1.f);
    pack_qk_kernel<<<(NHEAD * L_SEQ * 32) / 256, 256, 0, stream>>>(qkv_bf, qp, kp);
    transpose_v_kernel<<<(CH * L_SEQ) / 256, 256, 0, stream>>>(qkv_bf, vT);
    // scores_h = q_h k_h^T * scale + bias2d(+mask)
    for (int h = 0; h < NHEAD; ++h) {
      launch_gemm(stream, 4, 4, qp + (size_t)h * L_SEQ * 32, 32,
                  kp + (size_t)h * L_SEQ * 32, 32,
                  scores + (size_t)h * L_SEQ * L_SEQ, L_SEQ, nullptr,
                  nullptr, bias2d, L_SEQ, nullptr, 0,
                  L_SEQ, L_SEQ, 32, ATT_SCALE);
    }
    softmax_kernel<<<NHEAD * L_SEQ, 256, 0, stream>>>(scores, p_bf);
    // O_h = P_h @ V_h
    for (int h = 0; h < NHEAD; ++h) {
      launch_gemm(stream, 0, 1, p_bf + (size_t)h * L_SEQ * L_SEQ, L_SEQ,
                  vT + (size_t)h * HDIM * L_SEQ, L_SEQ,
                  o_f32 + h * HDIM, CH, o_bf + h * HDIM,
                  nullptr, nullptr, 0, nullptr, 0,
                  L_SEQ, HDIM, L_SEQ, 1.f);
    }
    // s = s + g_a * (o @ out_w[i] + out_b[i])
    launch_gemm(stream, 3, 4, o_bf, CH, ow_t + (size_t)i * CH * CH, CH,
                s_f32, CH, nullptr, out_b + (size_t)i * CH,
                s_f32, CH, mods + 2 * CH, 6 * CH, L_SEQ, CH, CH, 1.f);
    // yn = ln(s)*(1+sc_m)+sh_m
    ln_mod_kernel<<<L_SEQ / 8, 256, 0, stream>>>(s_f32, mods, 3 * CH, 4 * CH, xn_bf);
    // t = gelu(yn @ fc1_w[i] + fc1_b[i])  (bf16 only)
    launch_gemm(stream, 2, 4, xn_bf, CH, f1_t + (size_t)i * CH * CH, CH,
                nullptr, CH, t_bf, fc1_b + (size_t)i * CH,
                nullptr, 0, nullptr, 0, L_SEQ, CH, CH, 1.f);
    // s = s + g_m * (t @ fc2_w[i] + fc2_b[i])
    launch_gemm(stream, 3, 4, t_bf, CH, f2_t + (size_t)i * CH * CH, CH,
                s_f32, CH, nullptr, fc2_b + (size_t)i * CH,
                s_f32, CH, mods + 5 * CH, 6 * CH, L_SEQ, CH, CH, 1.f);
  }

  cvt4(stream, s_f32, s_bf, L_SEQ * CH);

  // plddt_logits = s @ ps_w + ps_b -> d_out[0 : 768*50]
  float* plddt = (float*)d_out;
  launch_gemm(stream, 0, 4, s_bf, CH, ps_t, CH, plddt, NB, nullptr,
              ps_b, nullptr, 0, nullptr, 0, L_SEQ, NB, CH, 1.f);

  // pde_logits -> d_out[768*50 : ]
  float* pde = plddt + (size_t)L_SEQ * NB;
  pde_kernel<<<L_SEQ, 256, 0, stream>>>(s_bf, s_f32, pz_w, pz_b, pde);
}
